// SimpleGraphModule_17617955848799
// MI455X (gfx1250) — matrix operand order repaired and checked by reading, not compiled, not run on hardware
//
#include <hip/hip_runtime.h>
#include <hip/hip_bf16.h>

typedef __attribute__((ext_vector_type(16))) _Float16 v16h;
typedef __attribute__((ext_vector_type(8)))  float    v8f;

#define BB   4
#define NN   1024
#define FF   128
#define HID  64
#define EMB  32
#define ATT  32

// ---------- WMMA fragment loaders (CDNA5 wave32 layouts, CDNA5 ISA) ----------
// A (16x32 f16): lane l -> row m = l&15, group g = l>>4 ; half h: K = g*8 + h (h<8), 16+g*8+(h-8)
__device__ __forceinline__ v16h load_frag_a_f32(const float* __restrict__ p, int ld, int lane) {
  int m = lane & 15, g = lane >> 4;
  const float* q = p + m * ld + g * 8;
  v16h a;
#pragma unroll
  for (int h = 0; h < 8; ++h) a[h] = (_Float16)q[h];
#pragma unroll
  for (int h = 0; h < 8; ++h) a[h + 8] = (_Float16)q[16 + h];
  return a;
}
__device__ __forceinline__ v16h load_frag_a_lds(const _Float16* __restrict__ p, int ld, int lane) {
  int m = lane & 15, g = lane >> 4;
  const _Float16* q = p + m * ld + g * 8;
  v16h a;
#pragma unroll
  for (int h = 0; h < 8; ++h) a[h] = q[h];
#pragma unroll
  for (int h = 0; h < 8; ++h) a[h + 8] = q[16 + h];
  return a;
}
// B (32x16 f16, K x N row-major source): lane l -> col n = l&15, g = l>>4 ; half h: K = g*8 + h (h<8), 16+g*8+(h-8)
__device__ __forceinline__ v16h load_frag_b_f32(const float* __restrict__ p, int ldn, int lane) {
  int n = lane & 15, g = lane >> 4;
  const float* q = p + (size_t)(g * 8) * ldn + n;
  v16h b;
#pragma unroll
  for (int h = 0; h < 8; ++h) b[h] = (_Float16)q[(size_t)h * ldn];
#pragma unroll
  for (int h = 0; h < 8; ++h) b[h + 8] = (_Float16)q[(size_t)(16 + h) * ldn];
  return b;
}

// ---------------- Kernel 1: node transform chain, all WMMA ----------------
// h = relu(x@W1+b1) ; e = h@W2+b2 ; a_i = e@Wa1[:32] ; a_j = e@Wa1[32:]
#define WAVES1 4
__global__ void __launch_bounds__(32 * WAVES1)
k_node(const float* __restrict__ x,  const float* __restrict__ W1, const float* __restrict__ b1,
       const float* __restrict__ W2, const float* __restrict__ b2, const float* __restrict__ Wa1,
       float* __restrict__ e_out, float* __restrict__ ai_out, float* __restrict__ aj_out) {
  __shared__ _Float16 stage[WAVES1][16 * HID];
  const int lane = threadIdx.x & 31;
  const int wave = threadIdx.x >> 5;
  const int tile = blockIdx.x * WAVES1 + wave;       // one 16-row node tile per wave
  const int b    = tile >> 6;                        // NN/16 == 64 tiles per batch
  const int row0 = (tile & 63) << 4;
  const int ln = lane & 15, g = lane >> 4;
  const float* xrow = x + ((size_t)b * NN + row0) * FF;
  _Float16* hbuf = stage[wave];

  // ---- step 1: h = relu(x @ W1 + b1) -> LDS f16 [16][64]
#pragma unroll
  for (int n0 = 0; n0 < HID; n0 += 16) {
    v8f acc = {};
#pragma unroll
    for (int k0 = 0; k0 < FF; k0 += 32) {
      v16h A = load_frag_a_f32(xrow + k0, FF, lane);
      v16h Bf = load_frag_b_f32(W1 + (size_t)k0 * HID + n0, HID, lane);
      acc = __builtin_amdgcn_wmma_f32_16x16x32_f16(false, A, false, Bf, (short)0, acc, false, false);
    }
    const float bias = b1[n0 + ln];
#pragma unroll
    for (int v = 0; v < 8; ++v) {
      int row = v + 8 * g;
      hbuf[row * HID + n0 + ln] = (_Float16)fmaxf(acc[v] + bias, 0.0f);
    }
  }
  __syncthreads();

  // ---- step 2: e = h @ W2 + b2   (K=64, N=32)
  v8f eacc[2];
#pragma unroll
  for (int t = 0; t < 2; ++t) {
    v8f acc = {};
#pragma unroll
    for (int k0 = 0; k0 < HID; k0 += 32) {
      v16h A = load_frag_a_lds(hbuf + k0, HID, lane);
      v16h Bf = load_frag_b_f32(W2 + (size_t)k0 * EMB + t * 16, EMB, lane);
      acc = __builtin_amdgcn_wmma_f32_16x16x32_f16(false, A, false, Bf, (short)0, acc, false, false);
    }
    const float bias = b2[t * 16 + ln];
#pragma unroll
    for (int v = 0; v < 8; ++v) acc[v] += bias;
    eacc[t] = acc;
  }
  __syncthreads();
  // write e: f32 to global (for final reduction) + f16 to LDS (A operand for step 3)
#pragma unroll
  for (int t = 0; t < 2; ++t) {
#pragma unroll
    for (int v = 0; v < 8; ++v) {
      int row = v + 8 * g, col = t * 16 + ln;
      e_out[((size_t)b * NN + row0 + row) * EMB + col] = eacc[t][v];
      hbuf[row * EMB + col] = (_Float16)eacc[t][v];
    }
  }
  __syncthreads();

  // ---- step 3: a_i = e @ Wa1[:EMB] ; a_j = e @ Wa1[EMB:]   (K=32 == one WMMA)
  v16h Ae = load_frag_a_lds(hbuf, EMB, lane);
#pragma unroll
  for (int half = 0; half < 2; ++half) {
    const float* Wbase = Wa1 + (size_t)half * EMB * ATT;   // top rows / bottom rows
    float* dst = half ? aj_out : ai_out;
#pragma unroll
    for (int t = 0; t < 2; ++t) {
      v8f acc = {};
      v16h Bf = load_frag_b_f32(Wbase + t * 16, ATT, lane);
      acc = __builtin_amdgcn_wmma_f32_16x16x32_f16(false, Ae, false, Bf, (short)0, acc, false, false);
#pragma unroll
      for (int v = 0; v < 8; ++v) {
        int row = v + 8 * g, col = t * 16 + ln;
        dst[((size_t)b * NN + row0 + row) * ATT + col] = acc[v];
      }
    }
  }
}

// ---------------- Kernel 2: fused edge stage + column sum ----------------
// s[b,j] = sum_i sigmoid(relu(a_i + a_j + ba1) . Wa2 + ba2) * adj[i,j]
#define JT 256
#define IT 64
__global__ void __launch_bounds__(JT)
k_edge(const float* __restrict__ ai, const float* __restrict__ aj,
       const float* __restrict__ adjm, const float* __restrict__ ba1,
       const float* __restrict__ Wa2, const float* __restrict__ ba2,
       float* __restrict__ s_out) {
  __shared__ float ai_lds[IT][ATT];
  const int b = blockIdx.y;
  const int j = blockIdx.x * JT + threadIdx.x;

  float ajb[ATT], w[ATT];
  const float* ajp = aj + ((size_t)b * NN + j) * ATT;
#pragma unroll
  for (int k = 0; k < ATT; ++k) { ajb[k] = ajp[k] + ba1[k]; w[k] = Wa2[k]; }
  const float bias2 = ba2[0];

  float s = 0.0f;
  for (int i0 = 0; i0 < NN; i0 += IT) {
    __syncthreads();
    for (int t = threadIdx.x; t < IT * ATT; t += JT)
      (&ai_lds[0][0])[t] = ai[((size_t)b * NN + i0) * ATT + t];
    __syncthreads();
#pragma unroll 4
    for (int ii = 0; ii < IT; ++ii) {
      float acc = bias2;
#pragma unroll
      for (int k = 0; k < ATT; ++k)
        acc = fmaf(fmaxf(ai_lds[ii][k] + ajb[k], 0.0f), w[k], acc);   // LDS broadcast reads
      float att = 1.0f / (1.0f + __expf(-acc));
      s = fmaf(att, adjm[(size_t)(i0 + ii) * NN + j], s);
    }
  }
  s_out[(size_t)b * NN + j] = s;
}

// ---------------- Kernel 3: out[b,d] = (1/N) * sum_j s[b,j] * e[b,j,d] ----------------
__global__ void __launch_bounds__(BB * EMB)
k_reduce(const float* __restrict__ s, const float* __restrict__ e, float* __restrict__ out) {
  const int b = threadIdx.x >> 5;   // EMB == 32 == wave width
  const int d = threadIdx.x & 31;
  float acc = 0.0f;
  for (int jj = 0; jj < NN; ++jj)
    acc = fmaf(s[(size_t)b * NN + jj], e[((size_t)b * NN + jj) * EMB + d], acc);
  out[b * EMB + d] = acc * (1.0f / NN);
}

extern "C" void kernel_launch(void* const* d_in, const int* in_sizes, int n_in,
                              void* d_out, int out_size, void* d_ws, size_t ws_size,
                              hipStream_t stream) {
  const float* x    = (const float*)d_in[0];
  const float* adjm = (const float*)d_in[1];
  const float* W1   = (const float*)d_in[2];
  const float* b1   = (const float*)d_in[3];
  const float* W2   = (const float*)d_in[4];
  const float* b2   = (const float*)d_in[5];
  const float* Wa1  = (const float*)d_in[6];
  const float* ba1  = (const float*)d_in[7];
  const float* Wa2  = (const float*)d_in[8];
  const float* ba2  = (const float*)d_in[9];
  float* out = (float*)d_out;

  // workspace layout (all f32): e | ai | aj | s
  float* e_ws  = (float*)d_ws;
  float* ai_ws = e_ws  + (size_t)BB * NN * EMB;
  float* aj_ws = ai_ws + (size_t)BB * NN * ATT;
  float* s_ws  = aj_ws + (size_t)BB * NN * ATT;

  // Kernel 1: 256 node tiles of 16 rows, 4 waves (128 threads) per block
  dim3 g1((BB * NN / 16) / WAVES1), t1(32 * WAVES1);
  k_node<<<g1, t1, 0, stream>>>(x, W1, b1, W2, b2, Wa1, e_ws, ai_ws, aj_ws);

  // Kernel 2: each thread owns one column j, loops all i with LDS-staged a_i tiles
  dim3 g2(NN / JT, BB), t2(JT);
  k_edge<<<g2, t2, 0, stream>>>(ai_ws, aj_ws, adjm, ba1, Wa2, ba2, s_ws);

  // Kernel 3: tiny weighted reduction
  k_reduce<<<1, BB * EMB, 0, stream>>>(s_ws, e_ws, out);
}
